// DeepseekV2MoE_64793876627498
// MI455X (gfx1250) — compile-verified, matrix operand order checked by reading, not run on hardware
//
#include <hip/hip_runtime.h>
#include <hip/hip_bf16.h>
#include <math.h>

#define T0_ 2048
#define D_  2048
#define E_  256
#define K_  8
#define G_  8
#define EG_ 32
#define R_  64
#define I_  1408
#define CAP_ (T0_*K_)   // 16384 dispatched rows total

typedef __bf16 v16bf __attribute__((ext_vector_type(16)));
typedef __bf16 bf8v  __attribute__((ext_vector_type(8)));
typedef float  v8f   __attribute__((ext_vector_type(8)));

static __device__ __forceinline__ v8f vzero8() {
  v8f z;
#pragma unroll
  for (int i = 0; i < 8; ++i) z[i] = 0.0f;
  return z;
}

static __device__ __forceinline__ v8f wmma_bf16(v16bf a, v16bf b, v8f c) {
  return __builtin_amdgcn_wmma_f32_16x16x32_bf16(false, a, false, b, (short)0, c,
                                                 false, false);
}

// ---- Fragment loader (ISA 7.12.2, wave32): two contiguous 16B runs/lane ----
static __device__ __forceinline__ v16bf load_frag(const __bf16* base, int ld, int k0) {
  int lane = threadIdx.x & 31;
  int sel = lane & 15, h = (lane >> 4) & 1;
  const __bf16* p = base + (size_t)sel * ld + k0 + h * 8;
  bf8v lo = *(const bf8v*)(p);
  bf8v hi = *(const bf8v*)(p + 16);
  v16bf r;
#pragma unroll
  for (int j = 0; j < 8; ++j) { r[j] = lo[j]; r[8 + j] = hi[j]; }
  return r;
}

// A-fragment with per-lane row gather (rows = tokens of this tile's rows).
static __device__ __forceinline__ v16bf load_frag_gather(const __bf16* mat, int ld,
                                                         const int* toks, int base,
                                                         int k0) {
  int lane = threadIdx.x & 31;
  int sel = lane & 15, h = (lane >> 4) & 1;
  int tk = toks[base + sel] & (T0_ - 1);      // clamp padded rows into range
  const __bf16* p = mat + (size_t)tk * ld + k0 + h * 8;
  bf8v lo = *(const bf8v*)(p);
  bf8v hi = *(const bf8v*)(p + 16);
  v16bf r;
#pragma unroll
  for (int j = 0; j < 8; ++j) { r[j] = lo[j]; r[8 + j] = hi[j]; }
  return r;
}

// ---- Utility kernels -------------------------------------------------------
__global__ void zero_f32_k(float* p, int n) {
  int i = blockIdx.x * 256 + threadIdx.x;
  if (i < n) p[i] = 0.0f;
}
__global__ void zero_i32_k(int* p, int n) {
  int i = blockIdx.x * 256 + threadIdx.x;
  if (i < n) p[i] = 0;
}
__global__ void cvt_bf16_k(const float* x, __bf16* y, int n) {
  int i = blockIdx.x * 256 + threadIdx.x;
  if (i < n) y[i] = (__bf16)x[i];
}
// Transpose+convert matrix blockIdx.y: fp32 [K x N] -> bf16 [N x K].
// k = blockIdx.x (no integer division); threads stride over N (coalesced read).
__global__ void pack_T_k(const float* __restrict__ in, __bf16* __restrict__ out,
                         int Kd, int Nd) {
  int k = blockIdx.x;
  const float* src = in + (size_t)blockIdx.y * Kd * Nd + (size_t)k * Nd;
  __bf16* dst = out + (size_t)blockIdx.y * Kd * Nd + k;
  for (int n = threadIdx.x; n < Nd; n += blockDim.x)
    dst[(size_t)n * Kd] = (__bf16)src[n];
}

// ---- Router GEMM: logits = hidden_bf16 @ w_gate ([T0 x D] @ [D x E]) -------
// grid (T0/64, E/64), block 128: wave = one 16-row tile, 4 N-subtiles.
__global__ void logits_gemm_k(const __bf16* __restrict__ h,
                              const __bf16* __restrict__ WgT, float* logits) {
  int wave = threadIdx.x >> 5, lane = threadIdx.x & 31;
  int row0 = blockIdx.x * 64 + wave * 16, col0 = blockIdx.y * 64;
  const __bf16* Ab = h + (size_t)row0 * D_;
  v8f acc[4];
#pragma unroll
  for (int s = 0; s < 4; ++s) acc[s] = vzero8();
  for (int kb = 0; kb < D_; kb += 32) {
    v16bf a = load_frag(Ab, D_, kb);
#pragma unroll
    for (int s = 0; s < 4; ++s)
      acc[s] = wmma_bf16(a, load_frag(WgT + (size_t)(col0 + s * 16) * D_, D_, kb), acc[s]);
  }
  int n = lane & 15, hh = (lane >> 4) & 1;
#pragma unroll
  for (int s = 0; s < 4; ++s)
#pragma unroll
    for (int r = 0; r < 8; ++r)
      logits[(size_t)(row0 + hh * 8 + r) * E_ + col0 + s * 16 + n] = acc[s][r];
}

// ---- Top-8 + renormalized softmax + per-EXPERT dispatch build --------------
__global__ void topk_k(const float* __restrict__ logits, int* cnt, int* etok,
                       float* ew) {
  int t = blockIdx.x;
  int tid = threadIdx.x;                      // 256 threads == E experts
  __shared__ float logit[E_];
  __shared__ float red[E_];
  __shared__ int redi[E_];
  __shared__ float topv[K_];
  __shared__ int topi[K_];
  logit[tid] = logits[(size_t)t * E_ + tid];
  __syncthreads();
  for (int k = 0; k < K_; ++k) {
    red[tid] = logit[tid];
    redi[tid] = tid;
    __syncthreads();
    for (int s = 128; s > 0; s >>= 1) {
      if (tid < s && red[tid + s] > red[tid]) {
        red[tid] = red[tid + s];
        redi[tid] = redi[tid + s];
      }
      __syncthreads();
    }
    if (tid == 0) {
      topv[k] = red[0];
      topi[k] = redi[0];
      logit[redi[0]] = -3.4e38f;
    }
    __syncthreads();
  }
  if (tid == 0) {
    float mx = topv[0], se = 0.0f, wv[K_];
#pragma unroll
    for (int k = 0; k < K_; ++k) { wv[k] = __expf(topv[k] - mx); se += wv[k]; }
#pragma unroll
    for (int k = 0; k < K_; ++k) {
      int e = topi[k];
      int slot = atomicAdd(&cnt[e], 1);       // per-expert cap == T0 (distinct ids)
      etok[e * T0_ + slot] = t;
      ew[e * T0_ + slot] = wv[k] / se;
    }
  }
}

// ---- H1[g] = hidden_bf16 @ A_{gate,up}[g]  ([T0 x D] @ [D x R]) ------------
// grid (T0/128, G, 2), block 128: wave = TWO 16-row tiles sharing 4 B-frags
// (12 b128 loads per 8 wmma instead of 10 per 4).
__global__ void h1_gemm_k(const __bf16* __restrict__ h, const __bf16* __restrict__ AgT,
                          const __bf16* __restrict__ AuT, __bf16* H1g, __bf16* H1u) {
  int g = blockIdx.y, f = blockIdx.z;
  const __bf16* Bp = (f == 0 ? AgT : AuT) + (size_t)g * R_ * D_;
  __bf16* O = (f == 0 ? H1g : H1u) + (size_t)g * T0_ * R_;
  int wave = threadIdx.x >> 5, lane = threadIdx.x & 31;
  int row0 = blockIdx.x * 128 + wave * 32;
  const __bf16* Ab0 = h + (size_t)row0 * D_;
  const __bf16* Ab1 = h + (size_t)(row0 + 16) * D_;
  v8f acc0[4], acc1[4];
#pragma unroll
  for (int s = 0; s < 4; ++s) { acc0[s] = vzero8(); acc1[s] = vzero8(); }
  for (int kb = 0; kb < D_; kb += 32) {
    v16bf a0 = load_frag(Ab0, D_, kb);
    v16bf a1 = load_frag(Ab1, D_, kb);
#pragma unroll
    for (int s = 0; s < 4; ++s) {
      v16bf b = load_frag(Bp + (size_t)(s * 16) * D_, D_, kb);
      acc0[s] = wmma_bf16(a0, b, acc0[s]);
      acc1[s] = wmma_bf16(a1, b, acc1[s]);
    }
  }
  int n = lane & 15, hh = (lane >> 4) & 1;
#pragma unroll
  for (int s = 0; s < 4; ++s)
#pragma unroll
    for (int r = 0; r < 8; ++r) {
      O[(size_t)(row0 + hh * 8 + r) * R_ + s * 16 + n] = (__bf16)acc0[s][r];
      O[(size_t)(row0 + 16 + hh * 8 + r) * R_ + s * 16 + n] = (__bf16)acc1[s][r];
    }
}

// ---- Prefix offsets & compaction (expert-sorted == group-sorted) -----------
__global__ void prefix_k(const int* cnt, int* off) {
  if (threadIdx.x == 0) {
    int s = 0;
    for (int e = 0; e < E_; ++e) { off[e] = s; s += cnt[e]; }
    off[E_] = s;                              // == CAP_
  }
}
__global__ void compact_k(const int* cnt, const int* off, const int* etok,
                          const float* ew, int* ftok, float* fw) {
  int e = blockIdx.x, nn = cnt[e], o = off[e];
  for (int i = threadIdx.x; i < nn; i += 256) {
    ftok[o + i] = etok[e * T0_ + i];
    fw[o + i] = ew[e * T0_ + i];
  }
}

// ---- Expert cores as WMMA: h2 = gather(H1) @ C[e]  (16-row expert tiles) ---
// grid (T0/16, E), block 32 (1 wave), early-exit past expert's row range.
__global__ void core_gu_k(const __bf16* __restrict__ H1g, const __bf16* __restrict__ H1u,
                          const __bf16* __restrict__ CgT, const __bf16* __restrict__ CuT,
                          const int* off, const int* ftok, __bf16* h2g, __bf16* h2u) {
  int e = blockIdx.y;
  int base = off[e] + blockIdx.x * 16;
  int end = off[e + 1];
  if (base >= end) return;
  int nv = end - base; if (nv > 16) nv = 16;
  int g = e >> 5;
  int lane = threadIdx.x & 31, n = lane & 15, hh = (lane >> 4) & 1;
  const __bf16* hg = H1g + (size_t)g * T0_ * R_;
  const __bf16* hu = H1u + (size_t)g * T0_ * R_;
  const __bf16* cg = CgT + (size_t)e * R_ * R_;
  const __bf16* cu = CuT + (size_t)e * R_ * R_;
  v16bf ag0 = load_frag_gather(hg, R_, ftok, base, 0);
  v16bf ag1 = load_frag_gather(hg, R_, ftok, base, 32);
  v16bf au0 = load_frag_gather(hu, R_, ftok, base, 0);
  v16bf au1 = load_frag_gather(hu, R_, ftok, base, 32);
#pragma unroll
  for (int s = 0; s < 4; ++s) {
    const __bf16* cgs = cg + (size_t)(s * 16) * R_;
    const __bf16* cus = cu + (size_t)(s * 16) * R_;
    v8f ga = wmma_bf16(ag1, load_frag(cgs, R_, 32),
                       wmma_bf16(ag0, load_frag(cgs, R_, 0), vzero8()));
    v8f ua = wmma_bf16(au1, load_frag(cus, R_, 32),
                       wmma_bf16(au0, load_frag(cus, R_, 0), vzero8()));
#pragma unroll
    for (int r = 0; r < 8; ++r) {
      int m = hh * 8 + r;
      if (m < nv) {
        h2g[(size_t)(base + m) * R_ + s * 16 + n] = (__bf16)ga[r];
        h2u[(size_t)(base + m) * R_ + s * 16 + n] = (__bf16)ua[r];
      }
    }
  }
}

// ---- Fused: go=h2g@Bg, uo=h2u@Bu, inter=silu(go)*uo (LDS), h1d+=inter@Ad ---
// grid (CAP/16, G), block 32. Expert-sorted rows are group-contiguous.
__global__ void fused_inter_down_k(const __bf16* __restrict__ h2g,
                                   const __bf16* __restrict__ h2u,
                                   const __bf16* __restrict__ BgT,
                                   const __bf16* __restrict__ BuT,
                                   const __bf16* __restrict__ AdT,
                                   const int* off, __bf16* h1d) {
  int g = blockIdx.y;
  int base = off[g * EG_] + blockIdx.x * 16;
  int end = off[(g + 1) * EG_];
  if (base >= end) return;
  int nv = end - base; if (nv > 16) nv = 16;
  int lane = threadIdx.x & 31, n = lane & 15, hh = (lane >> 4) & 1;
  const __bf16* bgT = BgT + (size_t)g * I_ * R_;   // [I][R]
  const __bf16* buT = BuT + (size_t)g * I_ * R_;   // [I][R]
  const __bf16* adT = AdT + (size_t)g * R_ * I_;   // [R][I]
  __shared__ __align__(16) __bf16 inter_l[16 * 64];

  v16bf ag0 = load_frag(h2g + (size_t)base * R_, R_, 0);
  v16bf ag1 = load_frag(h2g + (size_t)base * R_, R_, 32);
  v16bf au0 = load_frag(h2u + (size_t)base * R_, R_, 0);
  v16bf au1 = load_frag(h2u + (size_t)base * R_, R_, 32);
  v8f dacc[4];
#pragma unroll
  for (int s = 0; s < 4; ++s) dacc[s] = vzero8();

  for (int c = 0; c < I_; c += 64) {
    if (c + 64 < I_) {                        // global_prefetch_b8 of next chunk
      __builtin_prefetch(bgT + (size_t)(c + 64) * R_, 0, 1);
      __builtin_prefetch(buT + (size_t)(c + 64) * R_, 0, 1);
      __builtin_prefetch(adT + c + 64, 0, 1);
    }
#pragma unroll
    for (int sub = 0; sub < 4; ++sub) {
      const __bf16* bgc = bgT + (size_t)(c + sub * 16) * R_;
      const __bf16* buc = buT + (size_t)(c + sub * 16) * R_;
      v8f go = wmma_bf16(ag1, load_frag(bgc, R_, 32),
                         wmma_bf16(ag0, load_frag(bgc, R_, 0), vzero8()));
      v8f uo = wmma_bf16(au1, load_frag(buc, R_, 32),
                         wmma_bf16(au0, load_frag(buc, R_, 0), vzero8()));
#pragma unroll
      for (int r = 0; r < 8; ++r) {
        float x = go[r];
        float s = x / (1.0f + __expf(-x));    // silu
        inter_l[(hh * 8 + r) * 64 + sub * 16 + n] = (__bf16)(s * uo[r]);
      }
    }
    __syncthreads();                          // LDS b16 stores -> b128 A-frag reads
    v16bf ai0 = load_frag(inter_l, 64, 0);
    v16bf ai1 = load_frag(inter_l, 64, 32);
#pragma unroll
    for (int sub = 0; sub < 4; ++sub) {
      const __bf16* adc = adT + (size_t)(sub * 16) * I_ + c;
      dacc[sub] = wmma_bf16(ai0, load_frag(adc, I_, 0), dacc[sub]);
      dacc[sub] = wmma_bf16(ai1, load_frag(adc, I_, 32), dacc[sub]);
    }
    __syncthreads();
  }
#pragma unroll
  for (int sub = 0; sub < 4; ++sub)
#pragma unroll
    for (int r = 0; r < 8; ++r) {
      int m = hh * 8 + r;
      if (m < nv) h1d[(size_t)(base + m) * R_ + sub * 16 + n] = (__bf16)dacc[sub][r];
    }
}

// ---- Down core (WMMA) + weighted rank-space reduction into S[g,tok,:] ------
// grid (T0/16, E), block 32.
__global__ void core_down_scatter_k(const __bf16* __restrict__ h1d,
                                    const __bf16* __restrict__ CdT, const int* off,
                                    const int* ftok, const float* fw, float* S) {
  int e = blockIdx.y;
  int base = off[e] + blockIdx.x * 16;
  int end = off[e + 1];
  if (base >= end) return;
  int nv = end - base; if (nv > 16) nv = 16;
  int g = e >> 5;
  int lane = threadIdx.x & 31, n = lane & 15, hh = (lane >> 4) & 1;
  const __bf16* cd = CdT + (size_t)e * R_ * R_;
  v16bf a0 = load_frag(h1d + (size_t)base * R_, R_, 0);
  v16bf a1 = load_frag(h1d + (size_t)base * R_, R_, 32);
  int tokm[8]; float wm[8];
#pragma unroll
  for (int r = 0; r < 8; ++r) {
    int m = hh * 8 + r;
    if (m < nv) { tokm[r] = ftok[base + m]; wm[r] = fw[base + m]; }
    else        { tokm[r] = 0;             wm[r] = 0.0f; }
  }
#pragma unroll
  for (int s = 0; s < 4; ++s) {
    const __bf16* cds = cd + (size_t)(s * 16) * R_;
    v8f acc = wmma_bf16(a1, load_frag(cds, R_, 32),
                        wmma_bf16(a0, load_frag(cds, R_, 0), vzero8()));
#pragma unroll
    for (int r = 0; r < 8; ++r)
      if (wm[r] != 0.0f)
        atomicAdd(S + ((size_t)g * T0_ + tokm[r]) * R_ + s * 16 + n, wm[r] * acc[r]);
  }
}

// ---- out = sum_g Sb[g] @ B_down[g]  ([T0 x R] @ [R x D], 8 groups) ---------
// grid (T0/16, D/256), block 128: wave = 64 cols, A-frags reused 4x per group.
__global__ void out_gemm_k(const __bf16* __restrict__ Sb,
                           const __bf16* __restrict__ BdT, float* out) {
  int wave = threadIdx.x >> 5, lane = threadIdx.x & 31;
  int row0 = blockIdx.x * 16, col0 = blockIdx.y * 256 + wave * 64;
  v8f acc[4];
#pragma unroll
  for (int s = 0; s < 4; ++s) acc[s] = vzero8();
#pragma unroll
  for (int g = 0; g < G_; ++g) {
    const __bf16* Abase = Sb + ((size_t)g * T0_ + row0) * R_;
    v16bf a0 = load_frag(Abase, R_, 0);
    v16bf a1 = load_frag(Abase, R_, 32);
#pragma unroll
    for (int s = 0; s < 4; ++s) {
      const __bf16* Bbase = BdT + ((size_t)g * D_ + col0 + s * 16) * R_;
      acc[s] = wmma_bf16(a1, load_frag(Bbase, R_, 32),
                         wmma_bf16(a0, load_frag(Bbase, R_, 0), acc[s]));
    }
  }
  int n = lane & 15, hh = (lane >> 4) & 1;
#pragma unroll
  for (int s = 0; s < 4; ++s)
#pragma unroll
    for (int r = 0; r < 8; ++r)
      out[(size_t)(row0 + hh * 8 + r) * D_ + col0 + s * 16 + n] = acc[s][r];
}

// ---------------------------------------------------------------------------
extern "C" void kernel_launch(void* const* d_in, const int* in_sizes, int n_in,
                              void* d_out, int out_size, void* d_ws, size_t ws_size,
                              hipStream_t stream) {
  const float* hs = (const float*)d_in[0];
  const float* wg = (const float*)d_in[1];
  const float* Ag = (const float*)d_in[2];
  const float* Cg = (const float*)d_in[3];
  const float* Bg = (const float*)d_in[4];
  const float* Au = (const float*)d_in[5];
  const float* Cu = (const float*)d_in[6];
  const float* Bu = (const float*)d_in[7];
  const float* Ad = (const float*)d_in[8];
  const float* Cd = (const float*)d_in[9];
  const float* Bd = (const float*)d_in[10];
  float* out = (float*)d_out;

  char* w = (char*)d_ws;
  auto alloc = [&](size_t b) -> void* {
    void* p = (void*)w;
    w += (b + 255) & ~(size_t)255;
    return p;
  };
  __bf16* hbf   = (__bf16*)alloc((size_t)T0_ * D_ * 2);
  __bf16* WgT   = (__bf16*)alloc((size_t)E_ * D_ * 2);        // [E,D]
  __bf16* AgT   = (__bf16*)alloc((size_t)G_ * R_ * D_ * 2);   // [G,R,D]
  __bf16* AuT   = (__bf16*)alloc((size_t)G_ * R_ * D_ * 2);
  __bf16* BgT   = (__bf16*)alloc((size_t)G_ * I_ * R_ * 2);   // [G,I,R]
  __bf16* BuT   = (__bf16*)alloc((size_t)G_ * I_ * R_ * 2);
  __bf16* AdT   = (__bf16*)alloc((size_t)G_ * R_ * I_ * 2);   // [G,R,I]
  __bf16* BdT   = (__bf16*)alloc((size_t)G_ * D_ * R_ * 2);   // [G,D,R]
  __bf16* CgT   = (__bf16*)alloc((size_t)E_ * R_ * R_ * 2);   // [E,R,R] transposed
  __bf16* CuT   = (__bf16*)alloc((size_t)E_ * R_ * R_ * 2);
  __bf16* CdT   = (__bf16*)alloc((size_t)E_ * R_ * R_ * 2);
  __bf16* H1g   = (__bf16*)alloc((size_t)G_ * T0_ * R_ * 2);
  __bf16* H1u   = (__bf16*)alloc((size_t)G_ * T0_ * R_ * 2);
  float* logits = (float*)alloc((size_t)T0_ * E_ * 4);
  int* cnt      = (int*)alloc((size_t)E_ * 4);
  int* off      = (int*)alloc((size_t)(E_ + 1) * 4);
  int* etok     = (int*)alloc((size_t)E_ * T0_ * 4);
  float* ew     = (float*)alloc((size_t)E_ * T0_ * 4);
  int* ftok     = (int*)alloc((size_t)(CAP_ + 16) * 4);
  float* fw     = (float*)alloc((size_t)(CAP_ + 16) * 4);
  __bf16* h2g   = (__bf16*)alloc((size_t)(CAP_ + 16) * R_ * 2);
  __bf16* h2u   = (__bf16*)alloc((size_t)(CAP_ + 16) * R_ * 2);
  __bf16* h1d   = (__bf16*)alloc((size_t)(CAP_ + 16) * R_ * 2);
  float* S      = (float*)alloc((size_t)G_ * T0_ * R_ * 4);
  __bf16* Sb    = (__bf16*)alloc((size_t)G_ * T0_ * R_ * 2);

  zero_i32_k<<<1, 256, 0, stream>>>(cnt, E_);
  zero_f32_k<<<(G_ * T0_ * R_ + 255) / 256, 256, 0, stream>>>(S, G_ * T0_ * R_);
  cvt_bf16_k<<<(T0_ * D_ + 255) / 256, 256, 0, stream>>>(hs, hbf, T0_ * D_);

  // one-time bf16 transposed packing of all WMMA B-operands (no int division)
  pack_T_k<<<dim3(D_, 1), 256, 0, stream>>>(wg, WgT, D_, E_);
  pack_T_k<<<dim3(D_, G_), 64, 0, stream>>>(Ag, AgT, D_, R_);
  pack_T_k<<<dim3(D_, G_), 64, 0, stream>>>(Au, AuT, D_, R_);
  pack_T_k<<<dim3(R_, G_), 256, 0, stream>>>(Bg, BgT, R_, I_);
  pack_T_k<<<dim3(R_, G_), 256, 0, stream>>>(Bu, BuT, R_, I_);
  pack_T_k<<<dim3(I_, G_), 64, 0, stream>>>(Ad, AdT, I_, R_);
  pack_T_k<<<dim3(R_, G_), 256, 0, stream>>>(Bd, BdT, R_, D_);
  pack_T_k<<<dim3(R_, E_), 64, 0, stream>>>(Cg, CgT, R_, R_);
  pack_T_k<<<dim3(R_, E_), 64, 0, stream>>>(Cu, CuT, R_, R_);
  pack_T_k<<<dim3(R_, E_), 64, 0, stream>>>(Cd, CdT, R_, R_);

  logits_gemm_k<<<dim3(T0_ / 64, E_ / 64), 128, 0, stream>>>(hbf, WgT, logits);
  topk_k<<<T0_, E_, 0, stream>>>(logits, cnt, etok, ew);
  h1_gemm_k<<<dim3(T0_ / 128, G_, 2), 128, 0, stream>>>(hbf, AgT, AuT, H1g, H1u);
  prefix_k<<<1, 32, 0, stream>>>(cnt, off);
  compact_k<<<E_, 256, 0, stream>>>(cnt, off, etok, ew, ftok, fw);
  core_gu_k<<<dim3(T0_ / 16, E_), 32, 0, stream>>>(H1g, H1u, CgT, CuT, off, ftok, h2g, h2u);
  fused_inter_down_k<<<dim3(CAP_ / 16, G_), 32, 0, stream>>>(h2g, h2u, BgT, BuT, AdT, off, h1d);
  core_down_scatter_k<<<dim3(T0_ / 16, E_), 32, 0, stream>>>(h1d, CdT, off, ftok, fw, S);
  cvt_bf16_k<<<(G_ * T0_ * R_ + 255) / 256, 256, 0, stream>>>(S, Sb, G_ * T0_ * R_);
  out_gemm_k<<<dim3(T0_ / 16, D_ / 256), 128, 0, stream>>>(Sb, BdT, out);
}
